// ResidueBlock_25829933318893
// MI455X (gfx1250) — matrix operand order repaired and checked by reading, not compiled, not run on hardware
//
#include <hip/hip_runtime.h>
#include <hip/hip_bf16.h>

typedef __attribute__((ext_vector_type(8)))  _Float16 v8h;
typedef __attribute__((ext_vector_type(16))) _Float16 v16h;
typedef __attribute__((ext_vector_type(8)))  float    v8f;

#define POST_MEAN      0.29920671030107454f
#define INV_POST_STDV  (1.0f/0.664624264f)
#define R_HALF         0.70710678118654752f   /* sqrt(0.5) for both branches */

// ---------------------------------------------------------------------------
// helpers
// ---------------------------------------------------------------------------
__device__ __forceinline__ v16h cat16(v8h a, v8h b) {
    return __builtin_shufflevector(a, b, 0,1,2,3,4,5,6,7,8,9,10,11,12,13,14,15);
}

// split-f16 product: (Ah+Al)(Bh+Bl) ~= AhBh + AhBl + AlBh  (AlBl negligible)
__device__ __forceinline__ v8f wmma3(v16h ah, v16h al, v16h bh, v16h bl, v8f acc) {
    acc = __builtin_amdgcn_wmma_f32_16x16x32_f16(false, ah, false, bh, (short)0, acc, false, false);
    acc = __builtin_amdgcn_wmma_f32_16x16x32_f16(false, ah, false, bl, (short)0, acc, false, false);
    acc = __builtin_amdgcn_wmma_f32_16x16x32_f16(false, al, false, bh, (short)0, acc, false, false);
    return acc;
}

// raw 32-bit LDS byte address of a __shared__ object (AS3 pointers are 32-bit)
__device__ __forceinline__ uint32_t lds_addr32(const void* p) {
    return (uint32_t)(uintptr_t)(__attribute__((address_space(3))) const void*)p;
}

// CDNA5 async copy: LDS[vdst_lane] <- MEM[vaddr_lane], tracked by ASYNCcnt
__device__ __forceinline__ void async_load_b128(uint32_t lds, const void* gptr) {
    asm volatile("global_load_async_to_lds_b128 %0, %1, off"
                 :: "v"(lds), "v"(gptr)
                 : "memory");
}
__device__ __forceinline__ void wait_asynccnt0() {
    asm volatile("s_wait_asynccnt 0x0" ::: "memory");
}

// ---------------------------------------------------------------------------
// K0a: split conn (f32) into f16 hi/lo planes
// ---------------------------------------------------------------------------
__global__ __launch_bounds__(256) void split_conn_kernel(
    const float* __restrict__ src, _Float16* __restrict__ hi, _Float16* __restrict__ lo,
    long long n)
{
    long long i = ((long long)blockIdx.x * 256 + threadIdx.x) * 4;
    if (i + 3 < n) {
        float4 v = *(const float4*)(src + i);
        float  a[4] = {v.x, v.y, v.z, v.w};
        _Float16 h[4], l[4];
#pragma unroll
        for (int t = 0; t < 4; ++t) { h[t] = (_Float16)a[t]; l[t] = (_Float16)(a[t] - (float)h[t]); }
        *(ulong1*)(hi + i) = *(ulong1*)h;   // 8-byte stores
        *(ulong1*)(lo + i) = *(ulong1*)l;
    }
}

// ---------------------------------------------------------------------------
// K0b: pad + split weights, pad biases/prelus
//   slot 0: w1  83x64  -> 96x64     slot 1: w2 102x83 -> 128x96
//   slot 2: wm 128x102 -> 128x128   slot 3: ws 128x64 -> 128x64
// ---------------------------------------------------------------------------
__global__ __launch_bounds__(256) void prep_weights_kernel(
    const float* __restrict__ w1, const float* __restrict__ b1, const float* __restrict__ p1,
    const float* __restrict__ w2, const float* __restrict__ b2, const float* __restrict__ p2,
    const float* __restrict__ wm, const float* __restrict__ wsc,
    _Float16* __restrict__ W1hi, _Float16* __restrict__ W1lo,
    _Float16* __restrict__ W2hi, _Float16* __restrict__ W2lo,
    _Float16* __restrict__ WMhi, _Float16* __restrict__ WMlo,
    _Float16* __restrict__ WShi, _Float16* __restrict__ WSlo,
    float* __restrict__ b1p, float* __restrict__ p1p,
    float* __restrict__ b2p, float* __restrict__ p2p)
{
    const int slot = blockIdx.y;
    const int idx  = blockIdx.x * 256 + threadIdx.x;
    int Mact, Kact, Mpad, Kpad;
    const float* src; _Float16 *dhi, *dlo;
    switch (slot) {
        case 0:  Mact=83;  Kact=64;  Mpad=96;  Kpad=64;  src=w1;  dhi=W1hi; dlo=W1lo; break;
        case 1:  Mact=102; Kact=83;  Mpad=128; Kpad=96;  src=w2;  dhi=W2hi; dlo=W2lo; break;
        case 2:  Mact=128; Kact=102; Mpad=128; Kpad=128; src=wm;  dhi=WMhi; dlo=WMlo; break;
        default: Mact=128; Kact=64;  Mpad=128; Kpad=64;  src=wsc; dhi=WShi; dlo=WSlo; break;
    }
    if (idx < Mpad * Kpad) {
        int m = idx / Kpad, k = idx % Kpad;
        float v = (m < Mact && k < Kact) ? src[m * Kact + k] : 0.0f;
        _Float16 h = (_Float16)v;
        dhi[idx] = h;
        dlo[idx] = (_Float16)(v - (float)h);
    }
    if (blockIdx.x == 0) {
        int t = threadIdx.x;
        if (slot == 0 && t < 96)  { b1p[t] = (t < 83)  ? b1[t] : 0.f; p1p[t] = (t < 83)  ? p1[t] : 0.25f; }
        if (slot == 1 && t < 128) { b2p[t] = (t < 102) ? b2[t] : 0.f; p2p[t] = (t < 102) ? p2[t] : 0.25f; }
    }
}

// ---------------------------------------------------------------------------
// K1: fused channel chain, per (batch, 128-position tile)
// orientation-1 GEMM: A = weight planes (global, padded), B = activations in
// LDS stored transposed [pos][chan] (16-contiguous-K B-frags). Each wave owns
// a 16-position strip. Output per lane = 8 contiguous channels -> 16B stores.
// ---------------------------------------------------------------------------
template<int MTILES, int KTILES, bool CP, bool TO_LDS>
__device__ __forceinline__ void gemm_o1(
    const _Float16* __restrict__ Whi, const _Float16* __restrict__ Wlo, int Kpad,
    const float* __restrict__ bias, const float* __restrict__ prelu,
    const _Float16* shi, const _Float16* slo, int spitch,
    _Float16* dhi, _Float16* dlo, int dpitch,
    float* __restrict__ gbase,           // TO_LDS==false: fully-offset f32 row
    int nrow, int lmod, int lhalf)
{
    v8f acc[MTILES];
#pragma unroll
    for (int i = 0; i < MTILES; ++i) acc[i] = v8f{0,0,0,0,0,0,0,0};

#pragma unroll
    for (int kt = 0; kt < KTILES; ++kt) {
        const int k0 = kt * 32;
        const _Float16* sr_h = shi + nrow * spitch + k0 + lhalf * 16;
        const _Float16* sr_l = slo + nrow * spitch + k0 + lhalf * 16;
        v16h Bh = cat16(*(const v8h*)sr_h, *(const v8h*)(sr_h + 8));
        v16h Bl = cat16(*(const v8h*)sr_l, *(const v8h*)(sr_l + 8));
#pragma unroll
        for (int mt = 0; mt < MTILES; ++mt) {
            const _Float16* wr_h = Whi + (size_t)(mt*16 + lmod) * Kpad + k0;
            const _Float16* wr_l = Wlo + (size_t)(mt*16 + lmod) * Kpad + k0;
            v16h Ah = cat16(*(const v8h*)(wr_h + lhalf*16), *(const v8h*)(wr_h + lhalf*16 + 8));
            v16h Al = cat16(*(const v8h*)(wr_l + lhalf*16), *(const v8h*)(wr_l + lhalf*16 + 8));
            acc[mt] = wmma3(Ah, Al, Bh, Bl, acc[mt]);
        }
    }
#pragma unroll
    for (int mt = 0; mt < MTILES; ++mt) {
        const int m0 = mt*16 + lhalf*8;
        float vals[8];
#pragma unroll
        for (int r = 0; r < 8; ++r) {
            float v = acc[mt][r] + bias[m0 + r];
            if constexpr (CP) {
                float pv = prelu[m0 + r];
                v = (v >= 0.f) ? v : v * pv;
                v = (v - POST_MEAN) * INV_POST_STDV;
            }
            vals[r] = v;
        }
        if constexpr (TO_LDS) {
            v8h ph, pl;
#pragma unroll
            for (int r = 0; r < 8; ++r) {
                _Float16 h = (_Float16)vals[r];
                ph[r] = h; pl[r] = (_Float16)(vals[r] - (float)h);
            }
            *(v8h*)(dhi + nrow * dpitch + m0) = ph;
            *(v8h*)(dlo + nrow * dpitch + m0) = pl;
        } else {
            *(float4*)(gbase + m0)     = make_float4(vals[0], vals[1], vals[2], vals[3]);
            *(float4*)(gbase + m0 + 4) = make_float4(vals[4], vals[5], vals[6], vals[7]);
        }
    }
}

__global__ __launch_bounds__(256) void chain_kernel(
    const float* __restrict__ x,
    const _Float16* __restrict__ W1hi, const _Float16* __restrict__ W1lo,
    const _Float16* __restrict__ W2hi, const _Float16* __restrict__ W2lo,
    const _Float16* __restrict__ WMhi, const _Float16* __restrict__ WMlo,
    const _Float16* __restrict__ WShi, const _Float16* __restrict__ WSlo,
    const float* __restrict__ b1p, const float* __restrict__ p1p,
    const float* __restrict__ b2p, const float* __restrict__ p2p,
    const float* __restrict__ bm,  const float* __restrict__ bs,
    _Float16* __restrict__ hm_hi, _Float16* __restrict__ hm_lo,
    float* __restrict__ s_t)
{
    const int b     = blockIdx.y;
    const int n0    = blockIdx.x * 128;
    const int tid   = threadIdx.x;
    const int lane  = tid & 31;
    const int wid   = tid >> 5;
    const int lmod  = lane & 15;
    const int lhalf = lane >> 4;
    const int nw    = wid * 16;

    __shared__ _Float16 xt_hi[128][72],  xt_lo[128][72];
    __shared__ _Float16 h1_hi[128][104], h1_lo[128][104];
    __shared__ _Float16 h2_hi[128][136], h2_lo[128][136];

    { // stage x tile: transpose [c][n] -> [n][c], split hi/lo
        const float* xb = x + (size_t)b * 64 * 8192 + n0;
        for (int s = tid; s < 64 * 128; s += 256) {
            int c = s >> 7, n = s & 127;
            float v = xb[(size_t)c * 8192 + n];
            _Float16 h = (_Float16)v;
            xt_hi[n][c] = h;
            xt_lo[n][c] = (_Float16)(v - (float)h);
        }
    }
    __syncthreads();
    const int nrow = nw + lmod;

    // GEMM1: 96 x 128pos x 64  (cprelu)
    gemm_o1<6,2,true,true>(W1hi, W1lo, 64, b1p, p1p,
                           &xt_hi[0][0], &xt_lo[0][0], 72,
                           &h1_hi[0][0], &h1_lo[0][0], 104,
                           nullptr, nrow, lmod, lhalf);
    // GEMM2: 128 x 128pos x 96 (cprelu)
    gemm_o1<8,3,true,true>(W2hi, W2lo, 96, b2p, p2p,
                           &h1_hi[0][0], &h1_lo[0][0], 104,
                           &h2_hi[0][0], &h2_lo[0][0], 136,
                           nullptr, nrow, lmod, lhalf);
    // shortcut GEMM: 128 x 128pos x 64 -> s_t[i][bc] (f32, no activation)
    {
        float* gbase = s_t + (size_t)(n0 + nrow) * 2048 + (size_t)b * 128;
        gemm_o1<8,2,false,false>(WShi, WSlo, 64, bs, nullptr,
                                 &xt_hi[0][0], &xt_lo[0][0], 72,
                                 nullptr, nullptr, 0,
                                 gbase, nrow, lmod, lhalf);
    }
    __syncthreads();   // GEMM3 reads h2 rows written by all waves

    // GEMM3 (swapped orientation): A = h2^T (positions x K), B = wm^T.
    // D: M = position (contiguous runs), N = channel (fixed per lane)
    // -> hm[m][i] written with i-contiguous 16B stores (K2's B layout).
    {
        v8f acc[8];
#pragma unroll
        for (int i = 0; i < 8; ++i) acc[i] = v8f{0,0,0,0,0,0,0,0};
        const int mch = nw + lmod;   // this lane's output channel
#pragma unroll
        for (int kt = 0; kt < 4; ++kt) {
            const int k0 = kt * 32;
            const _Float16* wr_h = WMhi + (size_t)mch * 128 + k0 + lhalf * 16;
            const _Float16* wr_l = WMlo + (size_t)mch * 128 + k0 + lhalf * 16;
            v16h Bh = cat16(*(const v8h*)wr_h, *(const v8h*)(wr_h + 8));
            v16h Bl = cat16(*(const v8h*)wr_l, *(const v8h*)(wr_l + 8));
#pragma unroll
            for (int it = 0; it < 8; ++it) {
                const _Float16* ar_h = &h2_hi[it*16 + lmod][k0];
                const _Float16* ar_l = &h2_lo[it*16 + lmod][k0];
                v16h Ah = cat16(*(const v8h*)(ar_h + lhalf*16), *(const v8h*)(ar_h + lhalf*16 + 8));
                v16h Al = cat16(*(const v8h*)(ar_l + lhalf*16), *(const v8h*)(ar_l + lhalf*16 + 8));
                acc[it] = wmma3(Ah, Al, Bh, Bl, acc[it]);
            }
        }
        const float bv = bm[mch];
        _Float16* orow_h = hm_hi + ((size_t)b * 128 + mch) * 8192 + n0;
        _Float16* orow_l = hm_lo + ((size_t)b * 128 + mch) * 8192 + n0;
#pragma unroll
        for (int it = 0; it < 8; ++it) {
            v8h ph, pl;
#pragma unroll
            for (int r = 0; r < 8; ++r) {
                float v = acc[it][r] + bv;
                _Float16 h = (_Float16)v;
                ph[r] = h; pl[r] = (_Float16)(v - (float)h);
            }
            const int i0 = it*16 + lhalf*8;
            *(v8h*)(orow_h + i0) = ph;
            *(v8h*)(orow_l + i0) = pl;
        }
    }
}

// ---------------------------------------------------------------------------
// K2: mesh-pool GEMM  pool[o,m] = sum_i conn[o,i] * hm[m,i]
// A = conn (M=o), B = hm^T (N=m=b*128+c). Block 128x128, kc=32, 8 waves of
// 32(o) x 64(m), 24 WMMA per wave per K-chunk.
// Double-buffered LDS fed by CDNA5 async load-to-LDS (ASYNCcnt pipeline):
// chunk k+1 streams into buf^1 while chunk k feeds the WMMAs; one barrier
// per chunk. Epilogue: R_RES * cprelu(p3), coalesced 16B stores.
// ---------------------------------------------------------------------------
__global__ __launch_bounds__(256) void mesh_gemm_kernel(
    const _Float16* __restrict__ Chi, const _Float16* __restrict__ Clo,  // conn [4096][8192]
    const _Float16* __restrict__ Hhi, const _Float16* __restrict__ Hlo,  // hm   [2048][8192]
    const float* __restrict__ p3,
    float* __restrict__ out)                                             // [2048][4096]
{
    const int o0    = blockIdx.x * 128;
    const int m0    = blockIdx.y * 128;
    const int tid   = threadIdx.x;
    const int lane  = tid & 31;
    const int wid   = tid >> 5;
    const int lmod  = lane & 15;
    const int lhalf = lane >> 4;
    const int wo    = (wid & 3) * 32;    // wave o-offset
    const int wm_   = (wid >> 2) * 64;   // wave m-offset

    __shared__ _Float16 As_hi[2][128][40], As_lo[2][128][40];
    __shared__ _Float16 Bs_hi[2][128][40], Bs_lo[2][128][40];

    // this thread's staging slots (2 x 16B per plane per chunk)
    const int s0row = tid >> 2,            s0j = (tid & 3) * 8;
    const int s1row = (tid + 256) >> 2,    s1j = ((tid + 256) & 3) * 8;

    auto issue_chunk = [&](int k0, int buf) {
        size_t go0 = (size_t)(o0 + s0row) * 8192 + k0 + s0j;
        size_t gm0 = (size_t)(m0 + s0row) * 8192 + k0 + s0j;
        size_t go1 = (size_t)(o0 + s1row) * 8192 + k0 + s1j;
        size_t gm1 = (size_t)(m0 + s1row) * 8192 + k0 + s1j;
        async_load_b128(lds_addr32(&As_hi[buf][s0row][s0j]), Chi + go0);
        async_load_b128(lds_addr32(&As_lo[buf][s0row][s0j]), Clo + go0);
        async_load_b128(lds_addr32(&Bs_hi[buf][s0row][s0j]), Hhi + gm0);
        async_load_b128(lds_addr32(&Bs_lo[buf][s0row][s0j]), Hlo + gm0);
        async_load_b128(lds_addr32(&As_hi[buf][s1row][s1j]), Chi + go1);
        async_load_b128(lds_addr32(&As_lo[buf][s1row][s1j]), Clo + go1);
        async_load_b128(lds_addr32(&Bs_hi[buf][s1row][s1j]), Hhi + gm1);
        async_load_b128(lds_addr32(&Bs_lo[buf][s1row][s1j]), Hlo + gm1);
    };

    v8f acc[2][4];
#pragma unroll
    for (int f = 0; f < 2; ++f)
#pragma unroll
        for (int g = 0; g < 4; ++g) acc[f][g] = v8f{0,0,0,0,0,0,0,0};

    issue_chunk(0, 0);                                // prologue

    for (int k0 = 0, it = 0; k0 < 8192; k0 += 32, ++it) {
        const int cur = it & 1;
        wait_asynccnt0();        // this wave's chunk-k loads have landed in LDS
        __syncthreads();         // all waves' loads visible; buf[1-cur] free
        if (k0 + 32 < 8192) issue_chunk(k0 + 32, cur ^ 1);   // overlap compute

        v16h Af_h[2], Af_l[2], Bf_h[4], Bf_l[4];
#pragma unroll
        for (int f = 0; f < 2; ++f) {
            const _Float16* r_h = &As_hi[cur][wo + f*16 + lmod][lhalf*16];
            const _Float16* r_l = &As_lo[cur][wo + f*16 + lmod][lhalf*16];
            Af_h[f] = cat16(*(const v8h*)r_h, *(const v8h*)(r_h + 8));
            Af_l[f] = cat16(*(const v8h*)r_l, *(const v8h*)(r_l + 8));
        }
#pragma unroll
        for (int g = 0; g < 4; ++g) {
            const _Float16* r_h = &Bs_hi[cur][wm_ + g*16 + lmod][lhalf*16];
            const _Float16* r_l = &Bs_lo[cur][wm_ + g*16 + lmod][lhalf*16];
            Bf_h[g] = cat16(*(const v8h*)r_h, *(const v8h*)(r_h + 8));
            Bf_l[g] = cat16(*(const v8h*)r_l, *(const v8h*)(r_l + 8));
        }
#pragma unroll
        for (int f = 0; f < 2; ++f)
#pragma unroll
            for (int g = 0; g < 4; ++g)
                acc[f][g] = wmma3(Af_h[f], Af_l[f], Bf_h[g], Bf_l[g], acc[f][g]);
    }

#pragma unroll
    for (int g = 0; g < 4; ++g) {
        const int m  = m0 + wm_ + g*16 + lmod;
        const float pv = p3[m & 127];
        float* obase = out + (size_t)m * 4096 + o0 + wo;
#pragma unroll
        for (int f = 0; f < 2; ++f) {
            float vals[8];
#pragma unroll
            for (int r = 0; r < 8; ++r) {
                float v = acc[f][g][r];
                v = (v >= 0.f) ? v : v * pv;
                vals[r] = R_HALF * (v - POST_MEAN) * INV_POST_STDV;
            }
            float* p = obase + f*16 + lhalf*8;
            *(float4*)p       = make_float4(vals[0], vals[1], vals[2], vals[3]);
            *(float4*)(p + 4) = make_float4(vals[4], vals[5], vals[6], vals[7]);
        }
    }
}

// ---------------------------------------------------------------------------
// K3: shortcut: per conn row find argmax entries (ties summed, like the
// reference's >=max mask), gather s_t rows, out += R_SC * gathered.
// ---------------------------------------------------------------------------
__global__ __launch_bounds__(256) void shortcut_kernel(
    const float* __restrict__ conn,    // [4096][8192]
    const float* __restrict__ s_t,     // [8192][2048]
    float* __restrict__ out)           // [2048][4096]
{
    const int o   = blockIdx.x;
    const int tid = threadIdx.x;
    __shared__ float red[256];
    __shared__ int   idxs[16];
    __shared__ int   cnt;

    const float* row = conn + (size_t)o * 8192;
    float mx = -1e30f;
    for (int i = tid; i < 8192; i += 256) mx = fmaxf(mx, row[i]);
    red[tid] = mx;
    __syncthreads();
    for (int st = 128; st > 0; st >>= 1) {
        if (tid < st) red[tid] = fmaxf(red[tid], red[tid + st]);
        __syncthreads();
    }
    const float rmax = red[0];
    if (tid == 0) cnt = 0;
    __syncthreads();
    for (int i = tid; i < 8192; i += 256)
        if (row[i] >= rmax) { int j = atomicAdd(&cnt, 1); if (j < 16) idxs[j] = i; }
    __syncthreads();
    int c = cnt; if (c > 16) c = 16;

    float a[8] = {0,0,0,0,0,0,0,0};
    for (int j = 0; j < c; ++j) {
        const float* sp = s_t + (size_t)idxs[j] * 2048;
#pragma unroll
        for (int t = 0; t < 8; ++t) a[t] += sp[tid + t * 256];
    }
#pragma unroll
    for (int t = 0; t < 8; ++t) {
        size_t bc = (size_t)(tid + t * 256);
        out[bc * 4096 + o] += R_HALF * a[t];
    }
}

// ---------------------------------------------------------------------------
// launcher
// ---------------------------------------------------------------------------
extern "C" void kernel_launch(void* const* d_in, const int* in_sizes, int n_in,
                              void* d_out, int out_size, void* d_ws, size_t ws_size,
                              hipStream_t stream) {
    (void)in_sizes; (void)n_in; (void)out_size; (void)ws_size;
    const float* x    = (const float*)d_in[0];
    const float* conn = (const float*)d_in[1];
    const float* w1   = (const float*)d_in[2];
    const float* b1   = (const float*)d_in[3];
    const float* p1   = (const float*)d_in[4];
    const float* w2   = (const float*)d_in[5];
    const float* b2   = (const float*)d_in[6];
    const float* p2   = (const float*)d_in[7];
    const float* wm   = (const float*)d_in[8];
    const float* bm   = (const float*)d_in[9];
    const float* p3   = (const float*)d_in[10];
    const float* wsc  = (const float*)d_in[11];
    const float* bs   = (const float*)d_in[12];
    float* out = (float*)d_out;

    char* W = (char*)d_ws;
    const size_t MB = 1024ull * 1024ull;
    _Float16* conn_hi = (_Float16*)(W + 0);
    _Float16* conn_lo = (_Float16*)(W + 64 * MB);
    _Float16* hm_hi   = (_Float16*)(W + 128 * MB);
    _Float16* hm_lo   = (_Float16*)(W + 160 * MB);
    float*    s_t     = (float*)   (W + 192 * MB);
    char* P = W + 256 * MB;
    _Float16* W1hi = (_Float16*)(P + 0);
    _Float16* W1lo = (_Float16*)(P + 12288);
    _Float16* W2hi = (_Float16*)(P + 24576);
    _Float16* W2lo = (_Float16*)(P + 49152);
    _Float16* WMhi = (_Float16*)(P + 73728);
    _Float16* WMlo = (_Float16*)(P + 106496);
    _Float16* WShi = (_Float16*)(P + 139264);
    _Float16* WSlo = (_Float16*)(P + 155648);
    float*    b1p  = (float*)(P + 172032);
    float*    p1p  = (float*)(P + 172416);
    float*    b2p  = (float*)(P + 172800);
    float*    p2p  = (float*)(P + 173312);

    split_conn_kernel<<<32768, 256, 0, stream>>>(conn, conn_hi, conn_lo, 4096LL * 8192LL);
    prep_weights_kernel<<<dim3(64, 4), 256, 0, stream>>>(
        w1, b1, p1, w2, b2, p2, wm, wsc,
        W1hi, W1lo, W2hi, W2lo, WMhi, WMlo, WShi, WSlo, b1p, p1p, b2p, p2p);
    chain_kernel<<<dim3(64, 16), 256, 0, stream>>>(
        x, W1hi, W1lo, W2hi, W2lo, WMhi, WMlo, WShi, WSlo,
        b1p, p1p, b2p, p2p, bm, bs, hm_hi, hm_lo, s_t);
    mesh_gemm_kernel<<<dim3(32, 16), 256, 0, stream>>>(conn_hi, conn_lo, hm_hi, hm_lo, p3, out);
    shortcut_kernel<<<4096, 256, 0, stream>>>(conn, s_t, out);
}
